// CrosstalkTTS_8641474199750
// MI455X (gfx1250) — matrix operand order repaired and checked by reading, not compiled, not run on hardware
//
#include <hip/hip_runtime.h>

// ============================ model dims ============================
#define BB   8
#define TT   384
#define TMEL 1536
#define DD   256
#define HHD  4
#define M_ENC (BB * TT)     // 3072
#define M_DEC (BB * TMEL)   // 12288

typedef __attribute__((ext_vector_type(16))) _Float16 v16h;
typedef __attribute__((ext_vector_type(8)))  _Float16 v8h;
typedef __attribute__((ext_vector_type(8)))  float    v8f;

#define GF_BIAS  1
#define GF_RELU  2
#define GF_ACC   4
#define GF_SHIFT 8
#define GF_F16   16

static __device__ __forceinline__ v8f vzero8() {
  v8f v;
#pragma unroll
  for (int e = 0; e < 8; ++e) v[e] = 0.f;
  return v;
}

// ============================ generic WMMA GEMM ============================
// Y[M,N] = X[M,K](f32) @ W16^T, W16 [N,K] f16 row-major (pre-converted arena).
// NT = compile-time # of 16-col tiles per wave (kills runtime guards/spills).
// B tile = one contiguous v16h (2x b128) straight into the WMMA operand.
// A tile = 4x float4 loads converted component-wise (no local array punning).
template<int NT>
__global__ void k_gemm(const float* __restrict__ X, int ldx,
                       const _Float16* __restrict__ W16, int ldw,
                       const float* __restrict__ bias,
                       void* __restrict__ Yv, int ldy,
                       int colBase, int K, int rowShift, int Tlen, int flags)
{
  const int lane = threadIdx.x & 31;
  const int row0 = blockIdx.x * 16;
  const int col0 = colBase + blockIdx.y * (NT * 16);
  const int r    = row0 + (lane & 15);
  const int khi  = (lane >> 4) << 3;   // unused; A also reads at kb2 (contiguous K, 0 / 16)
  const int kb2  = (lane >> 4) << 4;   // B-matrix K split:      0 / 16

  const float* xrow;
  bool avalid = true;
  if (flags & GF_SHIFT) {
    int bidx = r / Tlen, t = r - bidx * Tlen;
    int ts = t + rowShift;
    avalid = (ts >= 0) && (ts < Tlen);
    xrow = X + (size_t)(bidx * Tlen + (avalid ? ts : t)) * ldx;
  } else {
    xrow = X + (size_t)r * ldx;
  }

  float* Yf = (float*)Yv;
  _Float16* Yh = (_Float16*)Yv;

  v8f acc[NT];
#pragma unroll
  for (int i = 0; i < NT; ++i) acc[i] = vzero8();
  if (flags & GF_ACC) {
#pragma unroll
    for (int i = 0; i < NT; ++i) {
      int cc = col0 + i * 16 + (lane & 15);
#pragma unroll
      for (int e = 0; e < 8; ++e) {
        int rr = row0 + ((lane < 16) ? e : e + 8);
        acc[i][e] = Yf[(size_t)rr * ldy + cc];
      }
    }
  }

  for (int k0 = 0; k0 < K; k0 += 32) {
    v16h a;
    if (avalid) {
      const float4* p0 = (const float4*)(xrow + k0 + kb2);
      const float4* p1 = (const float4*)(xrow + k0 + kb2 + 8);
      float4 f0 = p0[0], f1 = p0[1], f2 = p1[0], f3 = p1[1];
      a[0]  = (_Float16)f0.x; a[1]  = (_Float16)f0.y;
      a[2]  = (_Float16)f0.z; a[3]  = (_Float16)f0.w;
      a[4]  = (_Float16)f1.x; a[5]  = (_Float16)f1.y;
      a[6]  = (_Float16)f1.z; a[7]  = (_Float16)f1.w;
      a[8]  = (_Float16)f2.x; a[9]  = (_Float16)f2.y;
      a[10] = (_Float16)f2.z; a[11] = (_Float16)f2.w;
      a[12] = (_Float16)f3.x; a[13] = (_Float16)f3.y;
      a[14] = (_Float16)f3.z; a[15] = (_Float16)f3.w;
    } else {
#pragma unroll
      for (int e = 0; e < 16; ++e) a[e] = (_Float16)0.f;
    }
#pragma unroll
    for (int i = 0; i < NT; ++i) {
      int c = col0 + i * 16 + (lane & 15);
      v16h bm = *(const v16h*)(W16 + (size_t)c * ldw + k0 + kb2);
      acc[i] = __builtin_amdgcn_wmma_f32_16x16x32_f16(false, a, false, bm,
                                                      (short)0, acc[i], false, false);
    }
  }

#pragma unroll
  for (int i = 0; i < NT; ++i) {
    int cc = col0 + i * 16 + (lane & 15);
    float bv = (flags & GF_BIAS) ? bias[cc] : 0.f;
#pragma unroll
    for (int e = 0; e < 8; ++e) {
      int rr = row0 + ((lane < 16) ? e : e + 8);
      float v = acc[i][e] + bv;
      if (flags & GF_RELU) v = fmaxf(v, 0.f);
      if (flags & GF_F16) Yh[(size_t)rr * ldy + cc] = (_Float16)v;
      else                Yf[(size_t)rr * ldy + cc] = v;
    }
  }
}

// ============================ weight conversion ============================
__global__ void k_cvt(const float* __restrict__ s, _Float16* __restrict__ d, int n)
{
  int gid = blockIdx.x * 256 + threadIdx.x;
  if (gid < n) d[gid] = (_Float16)s[gid];
}
// [co][ci][3] -> 3 x [co][ci] f16
__global__ void k_cvt_conv(const float* __restrict__ cw, _Float16* __restrict__ d)
{
  int gid = blockIdx.x * 256 + threadIdx.x;     // 3*256*256
  int dt = gid >> 16, rem = gid & 65535;
  int co = rem >> 8, ci = rem & 255;
  d[gid] = (_Float16)cw[co * 768 + ci * 3 + dt];
}

// Q -> A-lane-order pack (scaled by 1/8), V -> [B,H,64,T] transpose.
// no perm within a 32-K block: k -> slot k (contiguous order)
__global__ void k_qvprep(const _Float16* __restrict__ qkv, _Float16* __restrict__ qp,
                         _Float16* __restrict__ vt, int T)
{
  int gid = blockIdx.x * 256 + threadIdx.x;     // over M*512
  int tok = gid >> 9, c = gid & 511;
  int b = tok / T, t = tok - b * T;
  if (c < 256) {
    int h = c >> 6, k = c & 63;
    int kk = k >> 5, w = k & 31, q = w >> 3;
    int slot = w + ((q == 1) ? 0 : (q == 2) ? 0 : 0);
    qp[((size_t)(b * HHD + h) * T + t) * 64 + kk * 32 + slot] =
        qkv[(size_t)tok * 768 + c] * (_Float16)0.125f;
  } else {
    int cc = c - 256;
    vt[((size_t)(b * HHD + (cc >> 6)) * 64 + (cc & 63)) * T + t] =
        qkv[(size_t)tok * 768 + 512 + cc];
  }
}

// ============================ flash attention ============================
// qp: Q packed in A order [B,H,T,64]; qkv: K read in place; vt: V^T [B,H,64,T].
// One wave / 16-query tile; online softmax over 32-key chunks; P staged in LDS
// in contiguous key order so the reload is a single 32B-aligned v16h DS read.
__global__ void k_attn(const _Float16* __restrict__ qkv, const _Float16* __restrict__ qp,
                       const _Float16* __restrict__ vt, float* __restrict__ out, int T)
{
  __shared__ alignas(32) _Float16 sPh[16 * 32];
  const int lane = threadIdx.x & 31;
  const int qt = blockIdx.x, hh = blockIdx.y, bb = blockIdx.z;
  const int kb2 = (lane >> 4) << 4;
  const _Float16* kbase = qkv + (size_t)bb * T * 768 + 256 + hh * 64;
  const int qrow = qt * 16 + (lane & 15);

  v16h qa[2];
  {
    const _Float16* q0 = qp + ((size_t)(bb * HHD + hh) * T + qrow) * 64 + kb2;
    qa[0] = *(const v16h*)(q0);
    qa[1] = *(const v16h*)(q0 + 32);
  }

  v8f o[4];
#pragma unroll
  for (int g = 0; g < 4; ++g) o[g] = vzero8();
  float m[8], l[8];
#pragma unroll
  for (int i = 0; i < 8; ++i) { m[i] = -1e30f; l[i] = 0.f; }

  for (int kb = 0; kb < T; kb += 32) {
    v8f s0 = vzero8(), s1 = vzero8();
#pragma unroll
    for (int kk = 0; kk < 2; ++kk) {
      v16h bm0 = *(const v16h*)(kbase + (size_t)(kb + (lane & 15)) * 768 + kk * 32 + kb2);
      s0 = __builtin_amdgcn_wmma_f32_16x16x32_f16(false, qa[kk], false, bm0, (short)0, s0, false, false);
      v16h bm1 = *(const v16h*)(kbase + (size_t)(kb + 16 + (lane & 15)) * 768 + kk * 32 + kb2);
      s1 = __builtin_amdgcn_wmma_f32_16x16x32_f16(false, qa[kk], false, bm1, (short)0, s1, false, false);
    }
    // online softmax; group-of-16 shfl reductions (a row spans 16 lanes)
#pragma unroll
    for (int i = 0; i < 8; ++i) {
      float cm = fmaxf(s0[i], s1[i]);
#pragma unroll
      for (int d = 1; d < 16; d <<= 1) cm = fmaxf(cm, __shfl_xor(cm, d, 32));
      float mn = fmaxf(m[i], cm);
      float sc = __expf(m[i] - mn);
      float p0 = __expf(s0[i] - mn);
      float p1 = __expf(s1[i] - mn);
      float rs = p0 + p1;
#pragma unroll
      for (int d = 1; d < 16; d <<= 1) rs += __shfl_xor(rs, d, 32);
      l[i] = l[i] * sc + rs;
      m[i] = mn;
#pragma unroll
      for (int g = 0; g < 4; ++g) o[g][i] *= sc;
      int rr = (lane < 16) ? i : i + 8;
      int cc = lane & 15;
      int basec = cc;                          // contiguous key slot
      sPh[rr * 32 + basec]     = (_Float16)p0;
      sPh[rr * 32 + basec + 16] = (_Float16)p1;
    }
    // reload P in A layout: one contiguous v16h per lane (same-wave DS in-order)
    v16h pa = *(const v16h*)(sPh + (lane & 15) * 32 + kb2);
    // O += P @ V  (V^T: keys contiguous)
#pragma unroll
    for (int g = 0; g < 4; ++g) {
      v16h vb = *(const v16h*)(vt + ((size_t)(bb * HHD + hh) * 64 + g * 16 + (lane & 15)) * T + kb + kb2);
      o[g] = __builtin_amdgcn_wmma_f32_16x16x32_f16(false, pa, false, vb, (short)0, o[g], false, false);
    }
  }

#pragma unroll
  for (int i = 0; i < 8; ++i) {
    float inv = 1.f / l[i];
#pragma unroll
    for (int g = 0; g < 4; ++g) o[g][i] *= inv;
  }
#pragma unroll
  for (int g = 0; g < 4; ++g) {
    int cc = hh * 64 + g * 16 + (lane & 15);
#pragma unroll
    for (int e = 0; e < 8; ++e) {
      int rr = qt * 16 + ((lane < 16) ? e : e + 8);
      out[(size_t)(bb * T + rr) * DD + cc] = o[g][e];
    }
  }
}

// ============================ rowwise kernels ============================
__global__ void k_ln(const float* __restrict__ x, const float* __restrict__ res,
                     const float* __restrict__ g, const float* __restrict__ bt,
                     const float* __restrict__ mask, float* __restrict__ y)
{
  const int tok = blockIdx.x;
  const int lane = threadIdx.x & 31;
  const float* xr = x + (size_t)tok * DD;
  float v[8]; float s = 0.f;
#pragma unroll
  for (int e = 0; e < 8; ++e) {
    float t = xr[lane + e * 32];
    if (res) t += res[(size_t)tok * DD + lane + e * 32];
    v[e] = t; s += t;
  }
#pragma unroll
  for (int d = 1; d < 32; d <<= 1) s += __shfl_xor(s, d, 32);
  float mean = s * (1.f / 256.f);
  float vs = 0.f;
#pragma unroll
  for (int e = 0; e < 8; ++e) { float d0 = v[e] - mean; vs += d0 * d0; }
#pragma unroll
  for (int d = 1; d < 32; d <<= 1) vs += __shfl_xor(vs, d, 32);
  float inv = rsqrtf(vs * (1.f / 256.f) + 1e-5f);
  float mk = mask ? mask[tok] : 1.f;
#pragma unroll
  for (int e = 0; e < 8; ++e) {
    int c = lane + e * 32;
    y[(size_t)tok * DD + c] = ((v[e] - mean) * inv * g[c] + bt[c]) * mk;
  }
}

// style-adaptive norm: (x-m)/(std_ddof1+1e-5) + style[b]
__global__ void k_san(const float* __restrict__ x, const float* __restrict__ style,
                      float* __restrict__ y, int Ttok)
{
  const int tok = blockIdx.x;
  const int lane = threadIdx.x & 31;
  const int b = tok / Ttok;
  const float* xr = x + (size_t)tok * DD;
  float v[8]; float s = 0.f;
#pragma unroll
  for (int e = 0; e < 8; ++e) { v[e] = xr[lane + e * 32]; s += v[e]; }
#pragma unroll
  for (int d = 1; d < 32; d <<= 1) s += __shfl_xor(s, d, 32);
  float mean = s * (1.f / 256.f);
  float vs = 0.f;
#pragma unroll
  for (int e = 0; e < 8; ++e) { float d0 = v[e] - mean; vs += d0 * d0; }
#pragma unroll
  for (int d = 1; d < 32; d <<= 1) vs += __shfl_xor(vs, d, 32);
  float sd = sqrtf(vs * (1.f / 255.f));        // ddof = 1
  float inv = 1.f / (sd + 1e-5f);
#pragma unroll
  for (int e = 0; e < 8; ++e) {
    int c = lane + e * 32;
    y[(size_t)tok * DD + c] = (v[e] - mean) * inv + style[b * DD + c];
  }
}

__global__ void k_embed(const int* __restrict__ ph, const float* __restrict__ emb,
                        float* __restrict__ X)
{
  int gid = blockIdx.x * 256 + threadIdx.x;
  int tok = gid >> 8, d = gid & 255;
  int t = tok % TT;
  int j = d >> 1;
  float ang = (float)t * __expf(-logf(10000.f) * (float)(2 * j) * (1.f / 256.f));
  float pe = (d & 1) ? __cosf(ang) : __sinf(ang);
  X[gid] = emb[(size_t)ph[tok] * DD + d] + pe;
}

__global__ void k_style(const int* __restrict__ spk_id, const int* __restrict__ emo_id,
                        const float* __restrict__ spk, const float* __restrict__ emo,
                        const float* __restrict__ w1, const float* __restrict__ b1,
                        const float* __restrict__ w2, const float* __restrict__ b2,
                        float* __restrict__ style)
{
  __shared__ float st[128], h1[256];
  int b = blockIdx.x, t = threadIdx.x;
  if (t < 64)        st[t] = spk[spk_id[b] * 64 + t];
  else if (t < 128)  st[t] = emo[emo_id[b] * 64 + (t - 64)];
  __syncthreads();
  float a = b1[t];
  for (int k = 0; k < 128; ++k) a += w1[t * 128 + k] * st[k];
  h1[t] = fmaxf(a, 0.f);
  __syncthreads();
  float o = b2[t];
  for (int k = 0; k < 256; ++k) o += w2[t * 256 + k] * h1[k];
  style[b * DD + t] = o;
}

__global__ void k_rowdot(const float* __restrict__ h, const float* __restrict__ lw,
                         const float* __restrict__ lb, float* __restrict__ y)
{
  int tok = blockIdx.x, lane = threadIdx.x & 31;
  float s = 0.f;
#pragma unroll
  for (int e = 0; e < 8; ++e) s += h[(size_t)tok * DD + lane + e * 32] * lw[lane + e * 32];
#pragma unroll
  for (int d = 1; d < 32; d <<= 1) s += __shfl_xor(s, d, 32);
  if (lane == 0) y[tok] = s + lb[0];
}

__global__ void k_cum(const int* __restrict__ dur, int* __restrict__ cum)
{
  int b = blockIdx.x;
  if (threadIdx.x == 0) {
    int s = 0;
    for (int t = 0; t < TT; ++t) { s += dur[b * TT + t]; cum[b * TT + t] = s; }
  }
}

__global__ void k_regidx(const int* __restrict__ cum, int* __restrict__ idx,
                         float* __restrict__ mmask)
{
  int gid = blockIdx.x * 256 + threadIdx.x;
  if (gid >= BB * TMEL) return;
  int b = gid / TMEL, pos = gid % TMEL;
  const int* c = cum + b * TT;
  int lo = 0, hi = TT;                 // searchsorted(side='right')
  while (lo < hi) { int mid = (lo + hi) >> 1; if (c[mid] <= pos) lo = mid + 1; else hi = mid; }
  idx[gid] = (lo > TT - 1) ? (TT - 1) : lo;
  mmask[gid] = (pos < c[TT - 1]) ? 1.f : 0.f;
}

__global__ void k_gather(const float* __restrict__ hid, const int* __restrict__ idx,
                         const float* __restrict__ mm, float* __restrict__ y)
{
  int gid = blockIdx.x * 256 + threadIdx.x;
  int tok = gid >> 8, d = gid & 255;
  int b = tok / TMEL;
  y[gid] = hid[((size_t)(b * TT + idx[tok])) * DD + d] * mm[tok];
}

__global__ void k_conv9add(float* __restrict__ X, const float* __restrict__ pred,
                           const float* __restrict__ w, const float* __restrict__ bias)
{
  int gid = blockIdx.x * 256 + threadIdx.x;
  int tok = gid >> 8, d = gid & 255;
  int b = tok / TMEL, t = tok % TMEL;
  float s = bias[d];
#pragma unroll
  for (int j = 0; j < 9; ++j) {
    int tt = t - 4 + j;
    if (tt >= 0 && tt < TMEL) s += w[d * 9 + j] * pred[b * TMEL + tt];
  }
  X[gid] += s;
}

__global__ void k_addstyle(float* __restrict__ X, const float* __restrict__ style)
{
  int gid = blockIdx.x * 256 + threadIdx.x;
  int tok = gid >> 8, d = gid & 255;
  X[gid] += style[(tok / TMEL) * DD + d];
}

// ============================ host orchestration ============================
namespace {

struct PCur {
  void* const* din; int idx;
  const float* flat; size_t off; bool single;
  const float* next(size_t n) {
    if (single) { const float* p = flat + off; off += n; return p; }
    (void)n; return (const float*)din[idx++];
  }
};
struct Layer {
  const float *b_in,*b_out,*b1,*b2,*ln1g,*ln1b,*ln2g,*ln2b;   // f32
  const float *w_in_f,*w_out_f,*w1_f,*w2_f;                   // f32 sources
  const _Float16 *w_in,*w_out,*w1,*w2;                        // f16 arena
};
struct VP {
  const float *cw0_f,*cw1_f;
  const _Float16 *c0,*c1;                                     // 3 taps each, tap i at +i*65536
  const float *cb0,*cb1,*lng0,*lng1,*lnb0,*lnb1,*lw,*lb;
};

Layer parseLayer(PCur& c) {
  Layer L;
  L.w_in_f = c.next(768*256);  L.b_in = c.next(768);
  L.w_out_f= c.next(256*256);  L.b_out= c.next(256);
  L.w1_f   = c.next(1024*256); L.b1   = c.next(1024);
  L.w2_f   = c.next(256*1024); L.b2   = c.next(256);
  L.ln1g = c.next(256); L.ln1b = c.next(256);
  L.ln2g = c.next(256); L.ln2b = c.next(256);
  return L;
}
VP parseVP(PCur& c) {
  VP v;
  v.cw0_f = c.next(256*256*3); v.cw1_f = c.next(256*256*3);
  v.cb0 = c.next(256); v.cb1 = c.next(256);
  v.lng0= c.next(256); v.lng1= c.next(256);
  v.lnb0= c.next(256); v.lnb1= c.next(256);
  v.lw  = c.next(256); v.lb  = c.next(1);
  return v;
}

inline void gemm(const float* X, int ldx, const _Float16* W16, int ldw,
                 const float* bias, void* Y, int ldy, int M, int N, int K,
                 int shift, int Tlen, int flags, hipStream_t s)
{
  int n4 = N >> 6;
  if (n4)
    k_gemm<4><<<dim3(M / 16, n4), 32, 0, s>>>(X, ldx, W16, ldw, bias, Y, ldy,
                                              0, K, shift, Tlen, flags);
  int rem = (N & 63) >> 4;
  if (rem == 1)
    k_gemm<1><<<dim3(M / 16, 1), 32, 0, s>>>(X, ldx, W16, ldw, bias, Y, ldy,
                                             n4 * 64, K, shift, Tlen, flags);
  else if (rem == 2)
    k_gemm<2><<<dim3(M / 16, 1), 32, 0, s>>>(X, ldx, W16, ldw, bias, Y, ldy,
                                             n4 * 64, K, shift, Tlen, flags);
  else if (rem == 3)
    k_gemm<3><<<dim3(M / 16, 1), 32, 0, s>>>(X, ldx, W16, ldw, bias, Y, ldy,
                                             n4 * 64, K, shift, Tlen, flags);
}

inline void run_block(const Layer& L, float* X0, float* X1, float* ATT, float* BIG,
                      _Float16* QKV16, _Float16* QP16, _Float16* VT16,
                      int Ttok, hipStream_t s)
{
  int M = BB * Ttok;
  gemm(X0, 256, L.w_in, 256, L.b_in, QKV16, 768, M, 768, 256, 0, 0, GF_BIAS | GF_F16, s);
  k_qvprep<<<M * 2, 256, 0, s>>>(QKV16, QP16, VT16, Ttok);
  k_attn<<<dim3(Ttok / 16, HHD, BB), 32, 0, s>>>(QKV16, QP16, VT16, ATT, Ttok);
  gemm(ATT, 256, L.w_out, 256, L.b_out, X1, 256, M, 256, 256, 0, 0, GF_BIAS, s);
  k_ln<<<M, 32, 0, s>>>(X0, X1, L.ln1g, L.ln1b, nullptr, X0);
  gemm(X0, 256, L.w1, 256, L.b1, BIG, 1024, M, 1024, 256, 0, 0, GF_BIAS | GF_RELU, s);
  gemm(BIG, 1024, L.w2, 1024, L.b2, X1, 256, M, 256, 1024, 0, 0, GF_BIAS, s);
  k_ln<<<M, 32, 0, s>>>(X0, X1, L.ln2g, L.ln2b, nullptr, X0);
}

// conv(k=3) blocks as 3 shifted GEMMs on repacked per-tap f16 weights.
inline void var_pred(const float* Xin, const float* mask, const VP& vp, float* pred,
                     float* VT0, float* VT1, int Ttok, hipStream_t s)
{
  int M = BB * Ttok;
  gemm(Xin, 256, vp.c0 + 0*65536, 256, vp.cb0, VT0, 256, M, 256, 256, -1, Ttok, GF_BIAS | GF_SHIFT, s);
  gemm(Xin, 256, vp.c0 + 1*65536, 256, nullptr, VT0, 256, M, 256, 256,  0, Ttok, GF_ACC  | GF_SHIFT, s);
  gemm(Xin, 256, vp.c0 + 2*65536, 256, nullptr, VT0, 256, M, 256, 256,  1, Ttok, GF_ACC  | GF_SHIFT | GF_RELU, s);
  k_ln<<<M, 32, 0, s>>>(VT0, nullptr, vp.lng0, vp.lnb0, mask, VT1);
  gemm(VT1, 256, vp.c1 + 0*65536, 256, vp.cb1, VT0, 256, M, 256, 256, -1, Ttok, GF_BIAS | GF_SHIFT, s);
  gemm(VT1, 256, vp.c1 + 1*65536, 256, nullptr, VT0, 256, M, 256, 256,  0, Ttok, GF_ACC  | GF_SHIFT, s);
  gemm(VT1, 256, vp.c1 + 2*65536, 256, nullptr, VT0, 256, M, 256, 256,  1, Ttok, GF_ACC  | GF_SHIFT | GF_RELU, s);
  k_ln<<<M, 32, 0, s>>>(VT0, nullptr, vp.lng1, vp.lnb1, mask, VT1);
  k_rowdot<<<M, 32, 0, s>>>(VT1, vp.lw, vp.lb, pred);
}

} // namespace

extern "C" void kernel_launch(void* const* d_in, const int* in_sizes, int n_in,
                              void* d_out, int out_size, void* d_ws, size_t ws_size,
                              hipStream_t stream)
{
  (void)in_sizes; (void)out_size; (void)ws_size;
  const int* ph     = (const int*)d_in[0];
  const int* spk_id = (const int*)d_in[1];
  const int* emo_id = (const int*)d_in[2];
  const int* dur    = (const int*)d_in[3];
  // d_in[4] = mel_len (compile-time 1536 here)

  PCur cur;
  cur.din = d_in; cur.idx = 5; cur.flat = nullptr; cur.off = 0; cur.single = false;
  if (n_in == 6) { cur.single = true; cur.flat = (const float*)d_in[5]; }

  // params in setup_inputs() insertion order
  const float* embed = cur.next(256 * 256);
  Layer enc[4]; for (int i = 0; i < 4; ++i) enc[i] = parseLayer(cur);
  const float* enc_ln_g = cur.next(256), *enc_ln_b = cur.next(256);
  const float* spk = cur.next(2 * 64), *emo = cur.next(5 * 64);
  const float* st_w1 = cur.next(256 * 128), *st_b1 = cur.next(256);
  const float* st_w2 = cur.next(256 * 256), *st_b2 = cur.next(256);
  VP dur_vp = parseVP(cur), pit_vp = parseVP(cur), ene_vp = parseVP(cur);
  const float* pe_w = cur.next(256 * 9), *pe_b = cur.next(256);
  const float* ee_w = cur.next(256 * 9), *ee_b = cur.next(256);
  Layer dec[4]; for (int i = 0; i < 4; ++i) dec[i] = parseLayer(cur);
  const float* dec_ln_g = cur.next(256), *dec_ln_b = cur.next(256);
  const float* mel_w_f = cur.next(80 * 256), *mel_b = cur.next(80);

  // ---- workspace layout ----
  float* ws  = (float*)d_ws;
  float* X0  = ws;
  float* X1  = X0  + (size_t)M_DEC * 256;
  float* ATT = X1  + (size_t)M_DEC * 256;
  float* HID = ATT + (size_t)M_DEC * 256;
  float* BIG = HID + (size_t)M_DEC * 256;      // M_DEC x 1024 f32
  float* SM  = BIG + (size_t)M_DEC * 1024;
  float* styleB = SM;                          // 8 x 256
  float* mmask  = SM + 2048;                   // 8 x 1536
  int*   cumB   = (int*)(mmask + BB * TMEL);   // 8 x 384
  int*   idxB   = cumB + BB * TT;              // 8 x 1536
  _Float16* QKV16 = (_Float16*)(idxB + BB * TMEL);       // M_DEC x 768 f16
  _Float16* QP16  = QKV16 + (size_t)M_DEC * 768;         // M_DEC x 256 f16 (packed Q)
  _Float16* VT16  = QP16  + (size_t)M_DEC * 256;         // B x H x 64 x TMEL f16
  _Float16* arena = VT16  + (size_t)M_DEC * 256;         // f16 weight arena
  float* VT0 = BIG;
  float* VT1 = BIG + (size_t)M_DEC * 256;

  // ---- weight conversion (once per launch; deterministic) ----
  _Float16* arp = arena;
  auto cvt = [&](const float* src, size_t n) -> const _Float16* {
    _Float16* dst = arp; arp += n;
    k_cvt<<<(int)((n + 255) / 256), 256, 0, stream>>>(src, dst, (int)n);
    return dst;
  };
  auto cvtconv = [&](const float* src) -> const _Float16* {
    _Float16* dst = arp; arp += 3 * 65536;
    k_cvt_conv<<<(3 * 65536) / 256, 256, 0, stream>>>(src, dst);
    return dst;
  };
  for (int i = 0; i < 4; ++i) {
    enc[i].w_in = cvt(enc[i].w_in_f, 768 * 256);
    enc[i].w_out= cvt(enc[i].w_out_f, 256 * 256);
    enc[i].w1   = cvt(enc[i].w1_f, 1024 * 256);
    enc[i].w2   = cvt(enc[i].w2_f, 256 * 1024);
  }
  for (int i = 0; i < 4; ++i) {
    dec[i].w_in = cvt(dec[i].w_in_f, 768 * 256);
    dec[i].w_out= cvt(dec[i].w_out_f, 256 * 256);
    dec[i].w1   = cvt(dec[i].w1_f, 1024 * 256);
    dec[i].w2   = cvt(dec[i].w2_f, 256 * 1024);
  }
  dur_vp.c0 = cvtconv(dur_vp.cw0_f); dur_vp.c1 = cvtconv(dur_vp.cw1_f);
  pit_vp.c0 = cvtconv(pit_vp.cw0_f); pit_vp.c1 = cvtconv(pit_vp.cw1_f);
  ene_vp.c0 = cvtconv(ene_vp.cw0_f); ene_vp.c1 = cvtconv(ene_vp.cw1_f);
  const _Float16* mel_w = cvt(mel_w_f, 80 * 256);

  float* out       = (float*)d_out;
  float* melOut    = out;                                    // 8*1536*80
  float* durOut    = out + (size_t)BB * TMEL * 80;           // 8*384
  float* pitchOut  = durOut + BB * TT;                       // 8*1536
  float* energyOut = pitchOut + BB * TMEL;                   // 8*1536

  // ---- text encoder ----
  k_embed<<<M_ENC, 256, 0, stream>>>(ph, embed, X0);
  for (int i = 0; i < 4; ++i)
    run_block(enc[i], X0, X1, ATT, BIG, QKV16, QP16, VT16, TT, stream);
  k_ln<<<M_ENC, 32, 0, stream>>>(X0, nullptr, enc_ln_g, enc_ln_b, nullptr, X1);

  // ---- style + style-adaptive norm ----
  k_style<<<BB, 256, 0, stream>>>(spk_id, emo_id, spk, emo, st_w1, st_b1, st_w2, st_b2, styleB);
  k_san<<<M_ENC, 32, 0, stream>>>(X1, styleB, HID, TT);

  // ---- duration predictor ----
  var_pred(HID, nullptr, dur_vp, durOut, VT0, VT1, TT, stream);

  // ---- length regulator ----
  k_cum<<<BB, 32, 0, stream>>>(dur, cumB);
  k_regidx<<<(BB * TMEL + 255) / 256, 256, 0, stream>>>(cumB, idxB, mmask);
  k_gather<<<M_DEC, 256, 0, stream>>>(HID, idxB, mmask, X0);

  // ---- pitch / energy ----
  var_pred(X0, mmask, pit_vp, pitchOut, VT0, VT1, TMEL, stream);
  k_conv9add<<<M_DEC, 256, 0, stream>>>(X0, pitchOut, pe_w, pe_b);
  var_pred(X0, mmask, ene_vp, energyOut, VT0, VT1, TMEL, stream);
  k_conv9add<<<M_DEC, 256, 0, stream>>>(X0, energyOut, ee_w, ee_b);

  // ---- mel decoder ----
  k_addstyle<<<M_DEC, 256, 0, stream>>>(X0, styleB);
  for (int i = 0; i < 4; ++i)
    run_block(dec[i], X0, X1, ATT, BIG, QKV16, QP16, VT16, TMEL, stream);
  k_ln<<<M_DEC, 32, 0, stream>>>(X0, nullptr, dec_ln_g, dec_ln_b, nullptr, X1);
  gemm(X1, 256, mel_w, 256, mel_b, melOut, 80, M_DEC, 80, 256, 0, 0, GF_BIAS, stream);
}
